// DMOR_31499290149350
// MI455X (gfx1250) — matrix operand order repaired and checked by reading, not compiled, not run on hardware
//
#include <hip/hip_runtime.h>
#include <hip/hip_bf16.h>
#include <math.h>

typedef __attribute__((ext_vector_type(16))) __bf16 v16bf;
typedef __attribute__((ext_vector_type(8)))  float  v8f;
typedef unsigned int __attribute__((ext_vector_type(4))) v4u;
typedef int  __attribute__((ext_vector_type(8))) v8i_t;
typedef int  __attribute__((ext_vector_type(4))) v4i_t;

#define BB   16
#define CC   32
#define HH   256
#define WW   256
#define TH   8
#define TW   16
#define HALO 2
#define TR   (TH + 2*HALO)   // 12
#define TC   (TW + 2*HALO)   // 20
#define HIDDEN 64
#define NFRAG 41             // 36 conv + 2 avg-identity + 2 identity + 1 router
#define WSWB  (NFRAG * 512 * 2)   // swizzled fragment image bytes = 41984

// ---------------------------------------------------------------------------
// Kernel 1: global average pool, one block per (b,c) plane -> gsum[b*C+c]
// ---------------------------------------------------------------------------
__global__ __launch_bounds__(256)
void gap_kernel(const float* __restrict__ x, float* __restrict__ gsum) {
    const int bc = blockIdx.x;                       // 0 .. B*C-1
    const float* p = x + (size_t)bc * (HH * WW);
    float s = 0.f;
    for (int i = threadIdx.x; i < HH * WW; i += 256) s += p[i];
    __shared__ float red[256];
    red[threadIdx.x] = s;
    __syncthreads();
    for (int off = 128; off > 0; off >>= 1) {
        if (threadIdx.x < off) red[threadIdx.x] += red[threadIdx.x + off];
        __syncthreads();
    }
    if (threadIdx.x == 0) gsum[bc] = red[0] * (1.0f / (HH * WW));
}

// ---------------------------------------------------------------------------
// Kernel 2: router MLP: gsum[B,C] -> relu(@gw1^T+gb1) -> @gw2^T+gb2 -> glog[B,4]
// ---------------------------------------------------------------------------
__global__ __launch_bounds__(1024)
void mlp_kernel(const float* __restrict__ gsum,
                const float* __restrict__ gw1, const float* __restrict__ gb1,
                const float* __restrict__ gw2, const float* __restrict__ gb2,
                float* __restrict__ glog) {
    __shared__ float h[BB * HIDDEN];
    const int t = threadIdx.x;                       // 1024 = 16*64
    const int b = t >> 6, j = t & 63;
    float s = gb1[j];
    #pragma unroll
    for (int c = 0; c < CC; ++c) s += gsum[b * CC + c] * gw1[j * CC + c];
    h[b * HIDDEN + j] = fmaxf(s, 0.f);
    __syncthreads();
    if (t < BB * 4) {
        const int b2 = t >> 2, n = t & 3;
        float s2 = gb2[n];
        #pragma unroll
        for (int jj = 0; jj < HIDDEN; ++jj) s2 += h[b2 * HIDDEN + jj] * gw2[n * HIDDEN + jj];
        glog[b2 * 4 + n] = s2;
    }
}

// ---------------------------------------------------------------------------
// Kernel 2b: build the swizzled A-fragment image ONCE in global memory.
//   frag  0..35 : conv weights ((conv*2+half)*9 + tap)
//   frag 36..37 : (1/9)*Identity (avgpool), half 0/1
//   frag 38..39 : Identity (identity op), half 0/1
//   frag 40     : spatial router sw, rows 4..15 zero
// element i of lane -> M = lane&15, K = i + 16*(lane>=16)
// (contiguous k order; the B loads in moe_main use the same order)
// ---------------------------------------------------------------------------
__global__ __launch_bounds__(256)
void weight_prep(const float* __restrict__ k1, const float* __restrict__ k2,
                 const float* __restrict__ sw, __bf16* __restrict__ wg) {
    const int e = blockIdx.x * 256 + threadIdx.x;
    if (e >= NFRAG * 512) return;
    const int i    = e & 15;
    const int lane = (e >> 4) & 31;
    const int frag = e >> 9;
    const int m = lane & 15;
    const int K = i + 16 * (lane >= 16);
    float val;
    if (frag < 36) {
        const int conv = frag / 18;
        const int rem  = frag % 18;
        const int half = rem / 9;
        const int tap  = rem % 9;
        const int co = half * 16 + m;
        const float* kk = conv ? k2 : k1;
        val = kk[(co * CC + K) * 9 + tap];
    } else if (frag == 36) {
        val = (K == m)      ? (1.0f / 9.0f) : 0.f;
    } else if (frag == 37) {
        val = (K == m + 16) ? (1.0f / 9.0f) : 0.f;
    } else if (frag == 38) {
        val = (K == m)      ? 1.0f : 0.f;
    } else if (frag == 39) {
        val = (K == m + 16) ? 1.0f : 0.f;
    } else {    // frag == 40
        val = (m < 4) ? sw[m * CC + K] : 0.f;
    }
    wg[e] = (__bf16)val;
}

// ---------------------------------------------------------------------------
// Kernel 3: fully fused. Every operator (conv d=1, conv d=2, avgpool, identity)
// and the spatial router run on the WMMA pipe against shared B fragments.
// A-fragments arrive via the Tensor Data Mover (tensor_load_to_lds),
// overlapped with the x-tile f32->bf16 staging loads.
// ---------------------------------------------------------------------------
__global__ __launch_bounds__(256)
void moe_main(const float* __restrict__ x,
              const __bf16* __restrict__ wg,
              const float* __restrict__ sb,
              const float* __restrict__ kb1, const float* __restrict__ kb2,
              const float* __restrict__ glog,
              float* __restrict__ out) {
    __shared__ __align__(64) __bf16 xt[TR * TC * CC];     // [pixel][ci] bf16 tile + halo
    __shared__ __align__(64) __bf16 wsw[NFRAG * 512];     // swizzled A-fragments [frag][lane][16]
    __shared__ float sbs[4];
    __shared__ float gls[4];

    const int tid = threadIdx.x;
    const int b  = blockIdx.z;
    const int h0 = blockIdx.y * TH;
    const int w0 = blockIdx.x * TW;

    // ---- wave 0: kick the TDM copy of the 41 KB fragment image into LDS ----
    if (tid < 32) {
        const unsigned long long ga = (unsigned long long)(size_t)wg;
        const unsigned int ldsoff = (unsigned int)(size_t)(void*)wsw;  // LDS offset = addr[31:0]
        const unsigned int nElem = WSWB / 8;                           // 8-byte elements

        v4u g0;
        g0.x = 1u;                                    // count=1, user descriptor
        g0.y = ldsoff;                                // lds_addr
        g0.z = (unsigned int)(ga & 0xffffffffu);      // global_addr[31:0]
        g0.w = (unsigned int)((ga >> 32) & 0x01ffffffu) | (2u << 30);  // addr[56:32] | type=2

        v8i_t g1;
        g1[0] = (int)(3u << 16);                      // data_size=3 (8B), mask=0
        g1[1] = (int)((nElem & 0xffffu) << 16);       // tensor_dim0[15:0]
        g1[2] = (int)(((nElem >> 16) & 0xffffu) | (1u << 16)); // dim0[31:16] | tensor_dim1=1
        g1[3] = (int)((nElem & 0xffffu) << 16);       // tile_dim0 = nElem
        g1[4] = 1;                                    // tile_dim1 = 1
        g1[5] = (int)nElem;                           // tensor_dim0_stride[31:0]
        g1[6] = 0;
        g1[7] = 0;

        v4i_t gz4 = {0, 0, 0, 0};
        v8i_t gz8 = {0, 0, 0, 0, 0, 0, 0, 0};
        __builtin_amdgcn_tensor_load_to_lds(g0, g1, gz4, gz4, gz8, 0);
    }

    // ---- stage per-op scalar constants ----
    if (tid < 4)      sbs[tid] = sb[tid];
    else if (tid < 8) gls[tid - 4] = glog[b * 4 + (tid - 4)];

    // ---- stage input tile, f32 -> bf16, zero-pad halo (SAME padding) ----
    const float* xb = x + (size_t)b * CC * HH * WW;
    for (int e = tid; e < CC * TR * TC; e += 256) {
        const int ci = e / (TR * TC);
        const int p  = e % (TR * TC);
        const int r = p / TC, c = p % TC;
        const int gr = h0 + r - HALO, gc = w0 + c - HALO;
        float v = 0.f;
        if (gr >= 0 && gr < HH && gc >= 0 && gc < WW)
            v = xb[(size_t)ci * HH * WW + (size_t)gr * WW + gc];
        xt[p * CC + ci] = (__bf16)v;
    }

    // ---- wait for the TDM transfer, then block barrier ----
    if (tid < 32) __builtin_amdgcn_s_wait_tensorcnt(0);
    __syncthreads();

    const int wv   = tid >> 5;       // wave id = tile row (0..7)
    const int lane = tid & 31;
    const int n    = lane & 15;      // pixel column within tile
    const int grp  = lane >> 4;

    v8f acc1a = {}, acc1b = {};      // conv d=1
    v8f acc2a = {}, acc2b = {};      // conv d=2
    v8f acc3a = {}, acc3b = {};      // avgpool

    const v16bf aAvg0 = *(const v16bf*)(wsw + 36 * 512 + lane * 16);
    const v16bf aAvg1 = *(const v16bf*)(wsw + 37 * 512 + lane * 16);

    #pragma unroll
    for (int tap = 0; tap < 9; ++tap) {
        const int dy = tap / 3 - 1, dx = tap % 3 - 1;
        {   // dilation-1 B fragment: feeds conv1 AND avgpool
            const int p = (HALO + wv + dy) * TC + (HALO + n + dx);
            const v16bf bf = *(const v16bf*)(xt + p * CC + grp * 16);
            const v16bf a0 = *(const v16bf*)(wsw + (0 * 9 + tap) * 512 + lane * 16);
            const v16bf a1 = *(const v16bf*)(wsw + (1 * 9 + tap) * 512 + lane * 16);
            acc1a = __builtin_amdgcn_wmma_f32_16x16x32_bf16(false, a0,   false, bf, (short)0, acc1a, false, false);
            acc1b = __builtin_amdgcn_wmma_f32_16x16x32_bf16(false, a1,   false, bf, (short)0, acc1b, false, false);
            acc3a = __builtin_amdgcn_wmma_f32_16x16x32_bf16(false, aAvg0, false, bf, (short)0, acc3a, false, false);
            acc3b = __builtin_amdgcn_wmma_f32_16x16x32_bf16(false, aAvg1, false, bf, (short)0, acc3b, false, false);
        }
        {   // dilation-2 B fragment: conv2
            const int p = (HALO + wv + 2 * dy) * TC + (HALO + n + 2 * dx);
            const v16bf bf = *(const v16bf*)(xt + p * CC + grp * 16);
            const v16bf a0 = *(const v16bf*)(wsw + (2 * 9 + tap) * 512 + lane * 16);
            const v16bf a1 = *(const v16bf*)(wsw + (3 * 9 + tap) * 512 + lane * 16);
            acc2a = __builtin_amdgcn_wmma_f32_16x16x32_bf16(false, a0, false, bf, (short)0, acc2a, false, false);
            acc2b = __builtin_amdgcn_wmma_f32_16x16x32_bf16(false, a1, false, bf, (short)0, acc2b, false, false);
        }
    }

    // ---- center-tap B fragment: identity op + spatial router on the WMMA pipe
    const v16bf bfc = *(const v16bf*)(xt + ((HALO + wv) * TC + (HALO + n)) * CC + grp * 16);
    v8f accIa = {}, accIb = {}, accS = {};
    {
        const v16bf aId0 = *(const v16bf*)(wsw + 38 * 512 + lane * 16);
        const v16bf aId1 = *(const v16bf*)(wsw + 39 * 512 + lane * 16);
        const v16bf aSw  = *(const v16bf*)(wsw + 40 * 512 + lane * 16);
        accIa = __builtin_amdgcn_wmma_f32_16x16x32_bf16(false, aId0, false, bfc, (short)0, accIa, false, false);
        accIb = __builtin_amdgcn_wmma_f32_16x16x32_bf16(false, aId1, false, bfc, (short)0, accIb, false, false);
        accS  = __builtin_amdgcn_wmma_f32_16x16x32_bf16(false, aSw,  false, bfc, (short)0, accS,  false, false);
    }

    // ---- routing for this lane's pixel (wv, n) ----
    // accS rows 0..3 live in VGPRs 0..3 of lanes 0..15 (N = lane). Broadcast to
    // the grp=1 lanes (same pixel, M rows 8..15 are zero) via wave shuffles.
    float logit[4];
    #pragma unroll
    for (int op = 0; op < 4; ++op)
        logit[op] = gls[op] + sbs[op] + __shfl(accS[op], n, 32);   // TEMP = 1.0

    const float m = fmaxf(fmaxf(logit[0], logit[1]), fmaxf(logit[2], logit[3]));
    float wgt[4]; float sum = 0.f;
    #pragma unroll
    for (int op = 0; op < 4; ++op) { wgt[op] = expf(logit[op] - m); sum += wgt[op]; }
    const float inv = 1.f / sum;
    #pragma unroll
    for (int op = 0; op < 4; ++op) wgt[op] *= inv;
    // top-2 sparsify + renorm
    int i0 = 0;
    #pragma unroll
    for (int op = 1; op < 4; ++op) if (wgt[op] > wgt[i0]) i0 = op;
    int i1 = (i0 == 0) ? 1 : 0;
    #pragma unroll
    for (int op = 0; op < 4; ++op) if (op != i0 && wgt[op] > wgt[i1]) i1 = op;
    const float keep = fmaxf(wgt[i0] + wgt[i1], 1e-6f);
    const float rk = 1.f / keep;
    #pragma unroll
    for (int op = 0; op < 4; ++op) wgt[op] = (op == i0 || op == i1) ? wgt[op] * rk : 0.f;

    // ---- blend operators (all already in 16x16 D layout) and store ----
    const int gr = h0 + wv, gc = w0 + n;
    float* ob = out + (size_t)b * CC * HH * WW + (size_t)gr * WW + gc;
    #pragma unroll
    for (int half = 0; half < 2; ++half) {
        const v8f a1 = half ? acc1b : acc1a;
        const v8f a2 = half ? acc2b : acc2a;
        const v8f a3 = half ? acc3b : acc3a;
        const v8f ai = half ? accIb : accIa;
        const int cobase = half * 16 + grp * 8;
        #pragma unroll
        for (int v = 0; v < 8; ++v) {
            const int co = cobase + v;                 // D layout: M = v + 8*grp
            const float c1 = a1[v] + kb1[co];
            const float c2 = a2[v] + kb2[co];
            ob[(size_t)co * HH * WW] =
                wgt[0] * ai[v] + wgt[1] * c1 + wgt[2] * c2 + wgt[3] * a3[v];
        }
    }
}

// ---------------------------------------------------------------------------
extern "C" void kernel_launch(void* const* d_in, const int* in_sizes, int n_in,
                              void* d_out, int out_size, void* d_ws, size_t ws_size,
                              hipStream_t stream) {
    const float* x   = (const float*)d_in[0];
    const float* gw1 = (const float*)d_in[1];
    const float* gb1 = (const float*)d_in[2];
    const float* gw2 = (const float*)d_in[3];
    const float* gb2 = (const float*)d_in[4];
    const float* sw  = (const float*)d_in[5];
    const float* sb  = (const float*)d_in[6];
    const float* k1  = (const float*)d_in[7];
    const float* kb1 = (const float*)d_in[8];
    const float* k2  = (const float*)d_in[9];
    const float* kb2 = (const float*)d_in[10];
    float* out = (float*)d_out;

    float* gsum    = (float*)d_ws;                       // B*C = 512 floats
    float* glogbuf = gsum + BB * CC;                     // B*4 = 64 floats
    __bf16* wswg   = (__bf16*)((char*)d_ws + 4096);      // 41984 B swizzled fragments

    gap_kernel<<<dim3(BB * CC), 256, 0, stream>>>(x, gsum);
    mlp_kernel<<<dim3(1), 1024, 0, stream>>>(gsum, gw1, gb1, gw2, gb2, glogbuf);
    weight_prep<<<dim3((NFRAG * 512 + 255) / 256), 256, 0, stream>>>(k1, k2, sw, wswg);
    moe_main<<<dim3(WW / TW, HH / TH, BB), 256, 0, stream>>>(
        x, wswg, sb, kb1, kb2, glogbuf, out);
}
